// BotRGCN_51024211476727
// MI455X (gfx1250) — matrix operand order repaired and checked by reading, not compiled, not run on hardware
//
#include <hip/hip_runtime.h>
#include <hip/hip_bf16.h>

typedef __attribute__((ext_vector_type(16))) _Float16 v16h;
typedef __attribute__((ext_vector_type(8)))  _Float16 v8h;
typedef __attribute__((ext_vector_type(4)))  _Float16 v4h;
typedef __attribute__((ext_vector_type(2)))  _Float16 v2h;
typedef __attribute__((ext_vector_type(8)))  float    v8f;
typedef int v4i_ __attribute__((vector_size(16)));   // int4 for async-LDS builtin

#define LRELU 1
#define APAD 40   // LDS row pitch in halves (bank spread, keeps 16B alignment)

#if defined(__has_builtin)
#if __has_builtin(__builtin_amdgcn_global_load_async_to_lds_b128) && \
    __has_builtin(__builtin_amdgcn_s_wait_asynccnt)
#define HAVE_ASYNC_LDS 1
#endif
#endif
#ifndef HAVE_ASYNC_LDS
#define HAVE_ASYNC_LDS 0
#endif

// ---------------------------------------------------------------------------
// Stage f32 weight [N,K] -> zero-padded f16 [Np, Kp] (Np,Kp multiples of 32).
// ---------------------------------------------------------------------------
__global__ void k_stagew(const float* __restrict__ W, int N, int K,
                         _Float16* __restrict__ W16, int Np, int Kp) {
  int i = blockIdx.x * blockDim.x + threadIdx.x;
  if (i >= Np * Kp) return;
  int n = i / Kp, k = i % Kp;
  float v = (n < N && k < K) ? W[(size_t)n * K + k] : 0.f;
  W16[i] = (_Float16)v;
}

// ---------------------------------------------------------------------------
// LDS-staged WMMA GEMM: C[M,N] = act( A[M,K] @ W16[N,K]^T + bias [+ C] )
// Block = 256 threads (8 waves) -> 64x32 C tile, BK = 32, K loop over Kpad.
// B fill is an unconditional 16B/lane copy of staged f16 weights
// (async global->LDS when the builtin exists). A fill fast path is
// manually unrolled (2x global b128 in flight) + prefetch of next K slice.
// ---------------------------------------------------------------------------
__global__ __launch_bounds__(256)
void k_gemm(const float* __restrict__ A, int lda,
            const _Float16* __restrict__ W16, int ldw,
            const float* __restrict__ bias,
            float* __restrict__ C, int ldc,
            int M, int N, int Kbound, int Kpad, int act, int accum) {
  __shared__ _Float16 sA[64 * APAD];
  __shared__ _Float16 sB[32 * APAD];

  const int tid  = threadIdx.x;
  const int lane = tid & 31;
  const int wid  = tid >> 5;
  const int wy   = wid & 3;
  const int wx   = wid >> 2;
  const int g    = lane >> 4;
  const int l    = lane & 15;
  const int m0   = blockIdx.y * 64;
  const int n0   = blockIdx.x * 32;
  const bool mFull = (m0 + 64 <= M);

  // per-thread A-fill coordinates (512 slots / 256 threads = 2 each)
  const int rowA0 = tid >> 3,          kqA0 = (tid & 7) << 2;
  const int rowA1 = (tid + 256) >> 3,  kqA1 = ((tid + 256) & 7) << 2;

  v8f acc = {};

  for (int k0 = 0; k0 < Kpad; k0 += 32) {
    if (mFull && (k0 + 32 <= Kbound)) {
      // ---- fast A fill: two b128 loads in flight, then cvt+store ---------
      const float* ap0 = A + (size_t)(m0 + rowA0) * lda + (k0 + kqA0);
      const float* ap1 = A + (size_t)(m0 + rowA1) * lda + (k0 + kqA1);
      float x0 = ap0[0], x1 = ap0[1], x2 = ap0[2], x3 = ap0[3];
      float y0 = ap1[0], y1 = ap1[1], y2 = ap1[2], y3 = ap1[3];
      __builtin_prefetch(ap0 + 32, 0, 0);   // next K slice (speculative)
      __builtin_prefetch(ap1 + 32, 0, 0);
      v4h p0, p1;
      p0[0] = (_Float16)x0; p0[1] = (_Float16)x1;
      p0[2] = (_Float16)x2; p0[3] = (_Float16)x3;
      p1[0] = (_Float16)y0; p1[1] = (_Float16)y1;
      p1[2] = (_Float16)y2; p1[3] = (_Float16)y3;
      *reinterpret_cast<v4h*>(&sA[rowA0 * APAD + kqA0]) = p0;
      *reinterpret_cast<v4h*>(&sA[rowA1 * APAD + kqA1]) = p1;
    } else {
      // ---- edge A fill: clamped address + zero select --------------------
      for (int t = tid; t < 64 * 16; t += 256) {
        int row = t >> 4;
        int kp  = (t & 15) << 1;
        int gr  = m0 + row;
        int gk  = k0 + kp;
        bool rOk = gr < M;
        const float* ar = A + (size_t)(rOk ? gr : 0) * lda;
        int c0 = gk     < Kbound ? gk     : 0;
        int c1 = gk + 1 < Kbound ? gk + 1 : 0;
        float a0 = ar[c0];
        float a1 = ar[c1];
        a0 = (rOk && gk     < Kbound) ? a0 : 0.f;
        a1 = (rOk && gk + 1 < Kbound) ? a1 : 0.f;
        v2h p; p[0] = (_Float16)a0; p[1] = (_Float16)a1;
        *reinterpret_cast<v2h*>(&sA[row * APAD + kp]) = p;
      }
    }
    // ---- B fill: unconditional 16B/lane copy of staged f16 weights -------
    if (tid < 128) {
      int n   = tid >> 2;
      int seg = (tid & 3) << 3;
      const _Float16* gsrc = W16 + (size_t)(n0 + n) * ldw + k0 + seg;
      _Float16* ldst = &sB[n * APAD + seg];
#if HAVE_ASYNC_LDS
      __builtin_amdgcn_global_load_async_to_lds_b128(
          (__attribute__((address_space(1))) v4i_*)gsrc,
          (__attribute__((address_space(3))) v4i_*)ldst,
          0, 0);
#else
      *reinterpret_cast<v8h*>(ldst) = *reinterpret_cast<const v8h*>(gsrc);
#endif
    }
#if HAVE_ASYNC_LDS
    __builtin_amdgcn_s_wait_asynccnt(0);
#endif
    __syncthreads();

    // ---- fragments from LDS (wide, unconditional) -------------------------
    const int rowHalf = (wy * 16 + l) * APAD;
    v8h alo = *reinterpret_cast<const v8h*>(&sA[rowHalf + 8 * g]);
    v8h ahi = *reinterpret_cast<const v8h*>(&sA[rowHalf + 16 + 8 * g]);
    v16h af = __builtin_shufflevector(alo, ahi, 0, 1, 2, 3, 4, 5, 6, 7,
                                      8, 9, 10, 11, 12, 13, 14, 15);
    const int colHalf = (wx * 16 + l) * APAD;
    v8h blo = *reinterpret_cast<const v8h*>(&sB[colHalf + 8 * g]);
    v8h bhi = *reinterpret_cast<const v8h*>(&sB[colHalf + 16 + 8 * g]);
    v16h bf = __builtin_shufflevector(blo, bhi, 0, 1, 2, 3, 4, 5, 6, 7,
                                      8, 9, 10, 11, 12, 13, 14, 15);

    acc = __builtin_amdgcn_wmma_f32_16x16x32_f16(
        false, af, false, bf, (short)0, acc, false, false);
    __syncthreads();
  }

  // ---- store: lanes0-15 col=l rows +0..7 ; lanes16-31 col=l rows +8..15 ---
  const int col = n0 + wx * 16 + l;
  if (col < N) {
    const float bv = bias[col];
#pragma unroll
    for (int r = 0; r < 8; ++r) {
      int row = m0 + wy * 16 + g * 8 + r;
      if (row < M) {
        float v = acc[r] + bv;
        if (accum) v += C[(size_t)row * ldc + col];
        if (act == LRELU) v = v > 0.f ? v : 0.01f * v;
        C[(size_t)row * ldc + col] = v;
      }
    }
  }
}

// ---------------------------------------------------------------------------
// Scatter-add: one wave per edge. ei: [2,E] int32 (src rows, then dst rows).
// ---------------------------------------------------------------------------
__global__ void k_scatter(const int* __restrict__ ei, int E,
                          const float* __restrict__ X, int ldx, int d,
                          float* __restrict__ sum, int lds,
                          float* __restrict__ cnt) {
  long long gid = (long long)blockIdx.x * blockDim.x + threadIdx.x;
  int e = (int)(gid >> 5);
  int lane = (int)(gid & 31);
  if (e >= E) return;
  int s = ei[e];
  int t = ei[E + e];
  if (lane == 0) atomicAdd(&cnt[t], 1.0f);
  for (int f = lane; f < d; f += 32)
    atomicAdd(&sum[(size_t)t * lds + f], X[(size_t)s * ldx + f]);
}

__global__ void k_mean(float* __restrict__ sum, int ld,
                       const float* __restrict__ cnt, int total, int d) {
  int i = blockIdx.x * blockDim.x + threadIdx.x;
  if (i >= total) return;
  int f = i % ld;
  int row = i / ld;
  if (f < d) sum[i] /= fmaxf(cnt[row], 1.0f);
}

__global__ void k_zero(float* __restrict__ p, int n) {
  int i = blockIdx.x * blockDim.x + threadIdx.x;
  if (i < n) p[i] = 0.0f;
}

// ---------------------------------------------------------------------------
// Row-wise (optional leaky-relu) + L2 normalize, in place. One wave per row.
// ld multiple of 32, <= 128; pad columns written as zero.
// ---------------------------------------------------------------------------
__global__ __launch_bounds__(32)
void k_norm(float* __restrict__ X, int ld, int d, int applyLrelu) {
  const int row = blockIdx.x;
  const int lane = threadIdx.x;
  const int nIter = ld >> 5;
  float vals[4];
  float ss = 0.f;
  for (int i = 0; i < nIter; ++i) {
    int f = lane + 32 * i;
    float v = (f < d) ? X[(size_t)row * ld + f] : 0.f;
    if (applyLrelu) v = v > 0.f ? v : 0.01f * v;
    vals[i] = v;
    ss += v * v;
  }
#pragma unroll
  for (int off = 16; off > 0; off >>= 1) ss += __shfl_xor(ss, off, 32);
  float inv = 1.f / fmaxf(sqrtf(ss), 1e-12f);
  for (int i = 0; i < nIter; ++i)
    X[(size_t)row * ld + lane + 32 * i] = vals[i] * inv;
}

// ---------------------------------------------------------------------------
extern "C" void kernel_launch(void* const* d_in, const int* in_sizes, int n_in,
                              void* d_out, int out_size, void* d_ws, size_t ws_size,
                              hipStream_t stream) {
  const int NU = 50000, NT = 150000, NL = 20000;
  const int H = 120;
  const int LDU = 128, LDT = 96, LDL = 64;

  // ---- inputs -------------------------------------------------------------
  const float* in_user_desc  = (const float*)d_in[0];
  const float* in_user_num   = (const float*)d_in[1];
  const float* in_user_cat   = (const float*)d_in[2];
  const float* in_tweet_num  = (const float*)d_in[3];
  const float* in_tweet_text = (const float*)d_in[4];
  const float* in_list_num   = (const float*)d_in[5];
  const float* in_list_desc  = (const float*)d_in[6];

  const int* EI[14];
  int        EN[14];
  for (int i = 0; i < 14; ++i) { EI[i] = (const int*)d_in[7 + i]; EN[i] = in_sizes[7 + i] / 2; }

  auto Wp = [&](int i) { return (const float*)d_in[21 + 2 * i]; };
  auto Bp = [&](int i) { return (const float*)d_in[22 + 2 * i]; };
  // 0 user_desc 1 user_num 2 user_cat 3 user_input 4 tweet_num 5 tweet_text
  // 6 tweet_input 7 list_num 8 list_desc 9 list_input 10..17 user_sage[0..7]
  // 18..20 tweet_sage 21..23 list_sage 24 user_self 25 user_neigh
  // 26 tweet_self 27 tweet_neigh 28 list_self 29 list_neigh 30 user_output 31 prediction

  static const int WN[32] = {40,40,40,120,30,60,90,30,30,60,
                             120,120,120,120,120,120,120,120,
                             90,90,90,60,60,60,
                             60,60,45,45,30,30,120,2};
  static const int WK[32] = {768,11,7,120,14,768,90,5,768,60,
                             120,120,90,90,90,60,60,60,
                             120,120,120,120,120,120,
                             120,120,90,90,60,60,120,120};

  // ---- workspace layout ---------------------------------------------------
  float* ws = (float*)d_ws;
  size_t o = 0;
  float* ux    = ws + o; o += (size_t)NU * LDU;
  float* ux2   = ws + o; o += (size_t)NU * LDU;
  float* tx    = ws + o; o += (size_t)NT * LDT;
  float* tx2   = ws + o; o += (size_t)NT * LDT;
  float* lx    = ws + o; o += (size_t)NL * LDL;
  float* lx2   = ws + o; o += (size_t)NL * LDL;
  float* big   = ws + o; o += (size_t)NT * LDU;   // cat buffers / per-rel mean buffer
  float* neigh = ws + o; o += (size_t)NT * LDT;   // per-block neighbor accumulator
  float* cnt   = ws + o; o += (size_t)NT;
  _Float16* w16base = (_Float16*)(ws + o);
  (void)ws_size; (void)n_in; (void)out_size;

  _Float16* w16p[32];
  int       wkp[32];
  {
    size_t wo = 0;
    for (int i = 0; i < 32; ++i) {
      int Np = (WN[i] + 31) & ~31;
      int Kp = (WK[i] + 31) & ~31;
      w16p[i] = w16base + wo;
      wkp[i]  = Kp;
      int tot = Np * Kp;
      k_stagew<<<(tot + 255) / 256, 256, 0, stream>>>(Wp(i), WN[i], WK[i],
                                                      w16p[i], Np, Kp);
      wo += (size_t)tot;
    }
  }

  // ---- helpers ------------------------------------------------------------
  auto gemm = [&](const float* A, int lda, int wi, float* C, int ldc,
                  int M, int N, int Kbound, int act, int accum) {
    dim3 grid((N + 31) / 32, (M + 63) / 64);
    k_gemm<<<grid, 256, 0, stream>>>(A, lda, w16p[wi], wkp[wi], Bp(wi),
                                     C, ldc, M, N, Kbound, wkp[wi], act, accum);
  };
  auto zero = [&](float* p, int n) {
    k_zero<<<(n + 255) / 256, 256, 0, stream>>>(p, n);
  };
  auto norml2 = [&](float* X, int ld, int n, int d, int lrelu) {
    k_norm<<<n, 32, 0, stream>>>(X, ld, d, lrelu);
  };
  // mean-aggregate X over edges into `big` (zero-padded), then sage GEMM
  auto sage = [&](int rel, const float* X, int ldx, int dIn, int nDst, int ldm,
                  int wi, float* out, int ldo, int dOut, int accum) {
    int E = EN[rel];
    zero(big, nDst * ldm);
    zero(cnt, nDst);
    long long thr = (long long)E * 32;
    k_scatter<<<(unsigned)((thr + 255) / 256), 256, 0, stream>>>(
        EI[rel], E, X, ldx, dIn, big, ldm, cnt);
    k_mean<<<(nDst * ldm + 255) / 256, 256, 0, stream>>>(big, ldm, cnt, nDst * ldm, dIn);
    gemm(big, ldm, wi, out, ldo, nDst, dOut, /*Kbound=*/ldm, 0, accum);
  };

  // ---- encoders -----------------------------------------------------------
  zero(big, NU * LDU);                                  // zero pads of cat buffer
  gemm(in_user_desc, 768, 0, big,      LDU, NU, 40, 768, LRELU, 0);
  gemm(in_user_num,   11, 1, big + 40, LDU, NU, 40,  11, LRELU, 0);
  gemm(in_user_cat,    7, 2, big + 80, LDU, NU, 40,   7, LRELU, 0);
  gemm(big, LDU, 3, ux, LDU, NU, H, LDU, LRELU, 0);
  norml2(ux, LDU, NU, H, 0);

  zero(big, NT * LDT);
  gemm(in_tweet_num,   14, 4, big,      LDT, NT, 30,  14, LRELU, 0);
  gemm(in_tweet_text, 768, 5, big + 30, LDT, NT, 60, 768, LRELU, 0);
  gemm(big, LDT, 6, tx, LDT, NT, 90, LDT, LRELU, 0);
  norml2(tx, LDT, NT, 90, 0);

  zero(big, NL * LDL);
  gemm(in_list_num,    5, 7, big,      LDL, NL, 30,   5, LRELU, 0);
  gemm(in_list_desc, 768, 8, big + 30, LDL, NL, 30, 768, LRELU, 0);
  gemm(big, LDL, 9, lx, LDL, NL, 60, LDL, LRELU, 0);
  norml2(lx, LDL, NL, 60, 0);

  // ---- user block (used twice) -------------------------------------------
  auto user_block = [&](const float* xu, const float* xt, const float* xl, float* out) {
    zero(neigh, NU * LDU);                              // keep pad columns zero
    sage(0, xu, LDU, 120, NU, LDU, 10, neigh, LDU, 120, 0);
    sage(1, xu, LDU, 120, NU, LDU, 11, neigh, LDU, 120, 1);
    sage(2, xt, LDT,  90, NU, LDT, 12, neigh, LDU, 120, 1);
    sage(3, xt, LDT,  90, NU, LDT, 13, neigh, LDU, 120, 1);
    sage(4, xt, LDT,  90, NU, LDT, 14, neigh, LDU, 120, 1);
    sage(5, xl, LDL,  60, NU, LDL, 15, neigh, LDU, 120, 1);
    sage(6, xl, LDL,  60, NU, LDL, 16, neigh, LDU, 120, 1);
    sage(7, xl, LDL,  60, NU, LDL, 17, neigh, LDU, 120, 1);
    gemm(xu,    LDU, 24, out,      LDU, NU, 60, LDU, 0, 0);
    gemm(neigh, LDU, 25, out + 60, LDU, NU, 60, LDU, 0, 0);
    norml2(out, LDU, NU, 120, 1);
  };

  // ---- round 1 ------------------------------------------------------------
  user_block(ux, tx, lx, ux2);

  zero(neigh, NT * LDT);
  sage(8,  ux, LDU, 120, NT, LDU, 18, neigh, LDT, 90, 0);
  sage(9,  ux, LDU, 120, NT, LDU, 19, neigh, LDT, 90, 1);
  sage(10, ux, LDU, 120, NT, LDU, 20, neigh, LDT, 90, 1);
  gemm(tx,    LDT, 26, tx2,      LDT, NT, 45, LDT, 0, 0);
  gemm(neigh, LDT, 27, tx2 + 45, LDT, NT, 45, LDT, 0, 0);
  norml2(tx2, LDT, NT, 90, 1);

  zero(neigh, NL * LDL);
  sage(11, ux, LDU, 120, NL, LDU, 21, neigh, LDL, 60, 0);
  sage(12, ux, LDU, 120, NL, LDU, 22, neigh, LDL, 60, 1);
  sage(13, ux, LDU, 120, NL, LDU, 23, neigh, LDL, 60, 1);
  gemm(lx,    LDL, 28, lx2,      LDL, NL, 30, LDL, 0, 0);
  gemm(neigh, LDL, 29, lx2 + 30, LDL, NL, 30, LDL, 0, 0);
  norml2(lx2, LDL, NL, 60, 1);

  // ---- round 2 (users only), ux buffer dead -> holds ux3 ------------------
  user_block(ux2, tx2, lx2, ux);

  // ---- head ---------------------------------------------------------------
  gemm(ux,  LDU, 30, ux2, LDU, NU, H, LDU, LRELU, 0);        // user_output
  gemm(ux2, LDU, 31, (float*)d_out, 2, NU, 2, LDU, 0, 0);    // prediction
}
